// EncDecAttention_11673721111124
// MI455X (gfx1250) — matrix operand order repaired and checked by reading, not compiled, not run on hardware
//
#include <hip/hip_runtime.h>
#include <hip/hip_bf16.h>
#include <stdint.h>

typedef __bf16 bf16_t;
typedef __bf16 v16bf __attribute__((ext_vector_type(16)));
typedef __bf16 v8bf  __attribute__((ext_vector_type(8)));
typedef float  v8f   __attribute__((ext_vector_type(8)));
typedef float  v4f   __attribute__((ext_vector_type(4)));

#define BB 4
#define NN 2048
#define MM 2048
#define CC 768
#define HH 12
#define DD 64

__device__ __forceinline__ v8f zero8() {
  v8f z;
#pragma unroll
  for (int i = 0; i < 8; ++i) z[i] = 0.0f;
  return z;
}

// Combine two 16-byte LDS/global runs into one 16x32 16-bit WMMA fragment.
__device__ __forceinline__ v16bf combine(v8bf lo, v8bf hi) {
  return __builtin_shufflevector(lo, hi, 0, 1, 2, 3, 4, 5, 6, 7,
                                 8, 9, 10, 11, 12, 13, 14, 15);
}
__device__ __forceinline__ v16bf load_frag(const bf16_t* p0, const bf16_t* p1) {
  return combine(*(const v8bf*)p0, *(const v8bf*)p1);
}
__device__ __forceinline__ void store8(bf16_t* dst, const float* s) {
  v8bf t;
#pragma unroll
  for (int i = 0; i < 8; ++i) t[i] = (bf16_t)s[i];
  *(v8bf*)dst = t;
}

// xor-lane shuffle via ds_swizzle (single instruction, wave32).
template <int XORM>
__device__ __forceinline__ float swz_xor(float v) {
  return __int_as_float(
      __builtin_amdgcn_ds_swizzle(__float_as_int(v), (XORM << 10) | 0x1F));
}

// ---------------------------------------------------------------------------
// Fused projection GEMM + per-head L2 norm.  OUT = X @ W^T.
// Block tile 128x64 (one head per col-tile); 8 waves x (16 rows x 64 cols).
// is_kv==0: q -> normalize + logit scale.  is_kv==1: col-block<12 -> k (norm),
// else -> v (plain).  bf16 out, coalesced via LDS repack.
// ---------------------------------------------------------------------------
__global__ __launch_bounds__(256)
void gemm_head_norm_kernel(const float* __restrict__ X, const float* __restrict__ W,
                           const float* __restrict__ logit_scale,
                           bf16_t* __restrict__ outA, bf16_t* __restrict__ outV,
                           int is_kv)
{
  __shared__ bf16_t As[128][40];   // pad 40 -> 16B-aligned frag runs, no bank conflicts
  __shared__ bf16_t Bs[64][40];
  __shared__ bf16_t Obuf[128][64]; // output repack

  const int tid  = threadIdx.x;
  const int wave = tid >> 5;
  const int lane = tid & 31;
  const int half = lane >> 4;
  const int l15  = lane & 15;
  const int r0   = blockIdx.x * 128;
  const int c0   = blockIdx.y * 64;

  v8f acc[4];
#pragma unroll
  for (int t = 0; t < 4; ++t) acc[t] = zero8();

  const int ar = tid >> 1, ac = (tid & 1) * 16;  // A stage: 16 elems/thread
  const int br = tid >> 2, bc = (tid & 3) * 8;   // W stage: 8 elems/thread

  for (int k0 = 0; k0 < CC; k0 += 32) {
    {
      const float* src = X + (size_t)(r0 + ar) * CC + k0 + ac;
      float f0[8], f1[8];
#pragma unroll
      for (int i = 0; i < 8; ++i) { f0[i] = src[i]; f1[i] = src[8 + i]; }
      store8(&As[ar][ac], f0);
      store8(&As[ar][ac + 8], f1);
    }
    {
      const float* src = W + (size_t)(c0 + br) * CC + k0 + bc;
      float f0[8];
#pragma unroll
      for (int i = 0; i < 8; ++i) f0[i] = src[i];
      store8(&Bs[br][bc], f0);
    }
    __syncthreads();

    const int row = wave * 16 + l15;
    v16bf a = load_frag(&As[row][8 * half], &As[row][16 + 8 * half]);
#pragma unroll
    for (int t = 0; t < 4; ++t) {
      const int c = 16 * t + l15;
      v16bf bfr = load_frag(&Bs[c][8 * half], &Bs[c][16 + 8 * half]);
      acc[t] = __builtin_amdgcn_wmma_f32_16x16x32_bf16(
          false, a, false, bfr, (short)0, acc[t], false, false);
    }
    __syncthreads();
  }

  // ---- per-row sum of squares: row r = e + 8*half of the wave strip ----
  float s[8];
#pragma unroll
  for (int e = 0; e < 8; ++e) {
    float a2 = 0.0f;
#pragma unroll
    for (int t = 0; t < 4; ++t) { float x = acc[t][e]; a2 += x * x; }
    float x = a2;
    x += swz_xor<1>(x); x += swz_xor<2>(x); x += swz_xor<4>(x); x += swz_xor<8>(x);
    s[e] = x;
  }

  const int hcol = blockIdx.y;
  bool do_norm, apply_scale;
  bf16_t* dst;
  int h;
  if (!is_kv)         { do_norm = true;  apply_scale = true;  dst = outA; h = hcol; }
  else if (hcol < HH) { do_norm = true;  apply_scale = false; dst = outA; h = hcol; }
  else                { do_norm = false; apply_scale = false; dst = outV; h = hcol - HH; }

  float sc = 1.0f;
  if (apply_scale) sc = __expf(fminf(logit_scale[h], 4.6051702f)); // log(1/0.01)

  float f[8];
#pragma unroll
  for (int e = 0; e < 8; ++e)
    f[e] = do_norm ? (sc / fmaxf(sqrtf(s[e]), 1e-12f)) : 1.0f;

  // repack into LDS, then contiguous 16B global stores (tile region is contiguous)
#pragma unroll
  for (int t = 0; t < 4; ++t)
#pragma unroll
    for (int e = 0; e < 8; ++e)
      Obuf[wave * 16 + e + 8 * half][16 * t + l15] = (bf16_t)(acc[t][e] * f[e]);
  __syncthreads();

  const int b = r0 >> 11;          // N == 2048
  const int nb = r0 & (NN - 1);
  bf16_t* ob = dst + (((size_t)b * HH + h) * NN + nb) * DD;
  const bf16_t* flat = &Obuf[0][0];
#pragma unroll
  for (int i = 0; i < 4; ++i) {
    int e0 = tid * 32 + i * 8;
    *(v8bf*)(ob + e0) = *(const v8bf*)(flat + e0);
  }
}

// ---------------------------------------------------------------------------
// Flash attention, transposed-S formulation, double-buffered K/V staging.
// K tiles staged with global_load_async_to_lds_b128 (ASYNCcnt path); V staged
// manually (transpose scatter).  Grid (N/128, H, B); 8 waves x 16 query rows.
// S^T = K_tile @ q^T -> softmax stats live per-lane (query = lane), and the
// resulting P^T accumulator IS the A-fragment layout for P@V (no shuffles).
// ---------------------------------------------------------------------------
__global__ __launch_bounds__(256)
void flash_attn_kernel(const bf16_t* __restrict__ qn, const bf16_t* __restrict__ kn,
                       const bf16_t* __restrict__ vn, const float* __restrict__ alibi,
                       const uint8_t* __restrict__ mask_p, float* __restrict__ xout)
{
  __shared__ bf16_t Ks[2][32][72];   // [buf][key][d]   (16B-aligned frag runs)
  __shared__ bf16_t Vt[2][64][40];   // [buf][d][key]   (transposed V)
  __shared__ float  cstat[8][16];    // per-wave row-broadcast scratch

  const int tid  = threadIdx.x;
  const int wave = tid >> 5;
  const int lane = tid & 31;
  const int half = lane >> 4;
  const int l15  = lane & 15;
  const int b = blockIdx.z, h = blockIdx.y;
  const int n0 = blockIdx.x * 128 + wave * 16;

  const bf16_t* kn_bh = kn + ((size_t)b * HH + h) * MM * DD;
  const bf16_t* v_bh  = vn + ((size_t)b * HH + h) * MM * DD;

  const int sr = tid >> 3, sd = (tid & 7) * 8;   // stage: 8 elems (16B) /thread

  auto stage = [&](int bufi, int ms) {
    // K tile: async DMA global -> LDS, one b128 per lane (tracked by ASYNCcnt)
    uint32_t lds_k = (uint32_t)(uintptr_t)&Ks[bufi][sr][sd];
    uint64_t g_k   = (uint64_t)(uintptr_t)(kn_bh + (size_t)(ms + sr) * DD + sd);
    asm volatile("global_load_async_to_lds_b128 %0, %1, off"
                 :: "v"(lds_k), "v"(g_k) : "memory");
    // V tile: load + transpose scatter into LDS
    v8bf v8 = *(const v8bf*)(v_bh + (size_t)(ms + sr) * DD + sd);
#pragma unroll
    for (int i = 0; i < 8; ++i) Vt[bufi][sd + i][sr] = v8[i];
    if (ms + 32 < MM)
      __builtin_prefetch(v_bh + (size_t)(ms + 32 + sr) * DD + sd, 0, 0);
  };

  // q B-fragments (K = d), live across the whole M loop
  const bf16_t* qrow = qn + (((size_t)b * HH + h) * NN + (n0 + l15)) * DD;
  v16bf bq0 = load_frag(qrow + 8 * half, qrow + 16 + 8 * half);
  v16bf bq1 = load_frag(qrow + 32 + 8 * half, qrow + 48 + 8 * half);

  v8f oacc[4];
#pragma unroll
  for (int t = 0; t < 4; ++t) oacc[t] = zero8();
  float mrow = -INFINITY;
  float lrow = 0.0f;

  stage(0, 0);
  for (int m0 = 0; m0 < MM; m0 += 32) {
    const int cur = (m0 >> 5) & 1;
    asm volatile("s_wait_asynccnt 0x0" ::: "memory");  // async K writes done
    __syncthreads();                       // staged tile visible; prev compute done
    if (m0 + 32 < MM) stage(cur ^ 1, m0 + 32);

    // ---- S^T (2 key subtiles x 16 queries) ----
    v8f st[2];
#pragma unroll
    for (int kt = 0; kt < 2; ++kt) {
      const int krow = 16 * kt + l15;
      v16bf a0 = load_frag(&Ks[cur][krow][8 * half], &Ks[cur][krow][16 + 8 * half]);
      v16bf a1 = load_frag(&Ks[cur][krow][32 + 8 * half], &Ks[cur][krow][48 + 8 * half]);
      v8f s0 = __builtin_amdgcn_wmma_f32_16x16x32_bf16(
          false, a0, false, bq0, (short)0, zero8(), false, false);
      st[kt] = __builtin_amdgcn_wmma_f32_16x16x32_bf16(
          false, a1, false, bq1, (short)0, s0, false, false);
    }

    // ---- alibi + padding mask (S^T layout: col = query = l15, row = key) ----
    const float* ab = alibi + ((size_t)h * NN + (n0 + l15)) * MM + m0;
    float vals[2][8];
#pragma unroll
    for (int kt = 0; kt < 2; ++kt) {
      const v4f* ab4 = (const v4f*)(ab + 16 * kt + 8 * half);
      v4f x0 = ab4[0], x1 = ab4[1];
      uint64_t mk8 = *(const uint64_t*)(mask_p + (size_t)b * MM + m0 + 16 * kt + 8 * half);
#pragma unroll
      for (int e = 0; e < 8; ++e) {
        float bias = (e < 4) ? x0[e] : x1[e & 3];
        bool msk = ((mk8 >> (8 * e)) & 0xffull) != 0;
        vals[kt][e] = msk ? -INFINITY : (st[kt][e] + bias);
      }
    }

    // ---- online softmax (per query column) ----
    float mloc = -INFINITY;
#pragma unroll
    for (int e = 0; e < 8; ++e)
      mloc = fmaxf(mloc, fmaxf(vals[0][e], vals[1][e]));
    mloc = fmaxf(mloc, swz_xor<16>(mloc));
    float mnew = fmaxf(mrow, mloc);
    float mm   = (mnew == -INFINITY) ? 0.0f : mnew;
    float corr = __expf(mrow - mm);
    float p0[8], p1[8], ssum = 0.0f;
#pragma unroll
    for (int e = 0; e < 8; ++e) {
      p0[e] = __expf(vals[0][e] - mm);
      p1[e] = __expf(vals[1][e] - mm);
      ssum += p0[e] + p1[e];
    }
    ssum += swz_xor<16>(ssum);
    lrow = lrow * corr + ssum;
    mrow = mnew;

    // ---- P^T is already the A-fragment for P@V ----
    v16bf ap;
#pragma unroll
    for (int e = 0; e < 8; ++e) { ap[e] = (bf16_t)p0[e]; ap[8 + e] = (bf16_t)p1[e]; }

    // rescale O: row r = e + 8*half needs corr of query r -> via per-wave LDS
    if (half == 0) cstat[wave][l15] = corr;
    float corrO[8];
#pragma unroll
    for (int e = 0; e < 8; ++e) corrO[e] = cstat[wave][8 * half + e];
#pragma unroll
    for (int t = 0; t < 4; ++t) {
#pragma unroll
      for (int e = 0; e < 8; ++e) oacc[t][e] *= corrO[e];
      const int dcol = 16 * t + l15;
      v16bf bv = load_frag(&Vt[cur][dcol][8 * half], &Vt[cur][dcol][16 + 8 * half]);
      oacc[t] = __builtin_amdgcn_wmma_f32_16x16x32_bf16(
          false, ap, false, bv, (short)0, oacc[t], false, false);
    }
  }

  // ---- finalize: divide by l, store x[B*N, C] at channel h*64+d ----
  if (half == 0) cstat[wave][l15] = lrow;
  float lr[8];
#pragma unroll
  for (int e = 0; e < 8; ++e) lr[e] = cstat[wave][8 * half + e];
#pragma unroll
  for (int t = 0; t < 4; ++t)
#pragma unroll
    for (int e = 0; e < 8; ++e) {
      int nrow = n0 + e + 8 * half;
      xout[(size_t)(b * NN + nrow) * CC + h * DD + 16 * t + l15] =
          oacc[t][e] / fmaxf(lr[e], 1e-30f);
    }
}

// ---------------------------------------------------------------------------
// Output projection: OUT = X @ W^T + bias (fp32 out).
// ---------------------------------------------------------------------------
__global__ __launch_bounds__(256)
void gemm_bias_kernel(const float* __restrict__ X, const float* __restrict__ W,
                      const float* __restrict__ bias, float* __restrict__ out)
{
  __shared__ bf16_t As[128][40];
  __shared__ bf16_t Bs[64][40];

  const int tid  = threadIdx.x;
  const int wave = tid >> 5;
  const int lane = tid & 31;
  const int half = lane >> 4;
  const int l15  = lane & 15;
  const int r0   = blockIdx.x * 128;
  const int c0   = blockIdx.y * 64;

  v8f acc[4];
#pragma unroll
  for (int t = 0; t < 4; ++t) acc[t] = zero8();

  const int ar = tid >> 1, ac = (tid & 1) * 16;
  const int br = tid >> 2, bc = (tid & 3) * 8;

  for (int k0 = 0; k0 < CC; k0 += 32) {
    {
      const float* src = X + (size_t)(r0 + ar) * CC + k0 + ac;
      float f0[8], f1[8];
#pragma unroll
      for (int i = 0; i < 8; ++i) { f0[i] = src[i]; f1[i] = src[8 + i]; }
      store8(&As[ar][ac], f0);
      store8(&As[ar][ac + 8], f1);
    }
    {
      const float* src = W + (size_t)(c0 + br) * CC + k0 + bc;
      float f0[8];
#pragma unroll
      for (int i = 0; i < 8; ++i) f0[i] = src[i];
      store8(&Bs[br][bc], f0);
    }
    __syncthreads();

    const int row = wave * 16 + l15;
    v16bf a = load_frag(&As[row][8 * half], &As[row][16 + 8 * half]);
#pragma unroll
    for (int t = 0; t < 4; ++t) {
      const int c = 16 * t + l15;
      v16bf bfr = load_frag(&Bs[c][8 * half], &Bs[c][16 + 8 * half]);
      acc[t] = __builtin_amdgcn_wmma_f32_16x16x32_bf16(
          false, a, false, bfr, (short)0, acc[t], false, false);
    }
    __syncthreads();
  }

#pragma unroll
  for (int t = 0; t < 4; ++t)
#pragma unroll
    for (int e = 0; e < 8; ++e) {
      int gr = r0 + wave * 16 + e + 8 * half;
      int col = c0 + 16 * t + l15;
      out[(size_t)gr * CC + col] = acc[t][e] + bias[col];
    }
}

// ---------------------------------------------------------------------------
extern "C" void kernel_launch(void* const* d_in, const int* in_sizes, int n_in,
                              void* d_out, int out_size, void* d_ws, size_t ws_size,
                              hipStream_t stream) {
  (void)in_sizes; (void)n_in; (void)out_size; (void)ws_size;
  const float*   q      = (const float*)d_in[0];
  const float*   kv     = (const float*)d_in[1];
  const uint8_t* pmask  = (const uint8_t*)d_in[2];
  const float*   alibi  = (const float*)d_in[3];
  const float*   q_w    = (const float*)d_in[4];
  const float*   kv_w   = (const float*)d_in[5];
  const float*   proj_w = (const float*)d_in[6];
  const float*   proj_b = (const float*)d_in[7];
  const float*   lscale = (const float*)d_in[8];

  char* wsb = (char*)d_ws;
  const size_t nh = (size_t)BB * HH * NN * DD;   // 6,291,456 elements per head buffer
  bf16_t* qn = (bf16_t*)(wsb);
  bf16_t* kn = (bf16_t*)(wsb + nh * 2);
  bf16_t* vb = (bf16_t*)(wsb + 2 * nh * 2);
  float*  xb = (float*)(wsb + 3 * nh * 2);

  gemm_head_norm_kernel<<<dim3(64, 12), 256, 0, stream>>>(q, q_w, lscale, qn, nullptr, 0);
  gemm_head_norm_kernel<<<dim3(64, 24), 256, 0, stream>>>(kv, kv_w, lscale, kn, vb, 1);
  flash_attn_kernel<<<dim3(16, 12, 4), 256, 0, stream>>>(qn, kn, vb, alibi, pmask, xb);
  gemm_bias_kernel<<<dim3(64, 12), 256, 0, stream>>>(xb, proj_w, proj_b, (float*)d_out);
}
